// NeighbourAssignment_72653666779329
// MI455X (gfx1250) — matrix operand order repaired and checked by reading, not compiled, not run on hardware
//
#include <hip/hip_runtime.h>

typedef __attribute__((ext_vector_type(16))) __bf16 v16bf;
typedef __attribute__((ext_vector_type(8)))  float  v8f;

#define WAVES   8
#define NB_OPS  34   // 17 column tiles (16 experts-cols + 1 Ws tile) * 2 K-steps

static __device__ __forceinline__ unsigned pack_bf16(float lo, float hi) {
    unsigned short a = __builtin_bit_cast(unsigned short, (__bf16)lo);
    unsigned short b = __builtin_bit_cast(unsigned short, (__bf16)hi);
    return (unsigned)a | ((unsigned)b << 16);
}

static __device__ __forceinline__ float f4get(const float4& v, int s) {
    return s == 0 ? v.x : s == 1 ? v.y : s == 2 ? v.z : v.w;
}

static __device__ __forceinline__ v8f wmma_bf16(v16bf a, v16bf b, v8f c) {
    // D = A(16x32 bf16) * B(32x16 bf16) + C(16x16 f32)
    return __builtin_amdgcn_wmma_f32_16x16x32_bf16(false, a, false, b, (short)0, c,
                                                   false, false);
}

// Build the A operand for one 16x32 bf16 tile from a gathered f32 row.
// Lane layout (ISA 7.12.2): lanes 0-15 hold K=klo..klo+7 and klo+16..klo+23,
// lanes 16-31 hold K=klo+8.. and klo+24.. (klo folded into pointer below).
static __device__ __forceinline__ v16bf make_a(const float* rowp, int klo) {
    float4 f0 = *(const float4*)(rowp + klo);
    float4 f1 = *(const float4*)(rowp + klo + 4);
    float4 f2 = *(const float4*)(rowp + klo + 16);
    float4 f3 = *(const float4*)(rowp + klo + 20);
    v16bf v;
    v[0]  = (__bf16)f0.x; v[1]  = (__bf16)f0.y; v[2]  = (__bf16)f0.z; v[3]  = (__bf16)f0.w;
    v[4]  = (__bf16)f1.x; v[5]  = (__bf16)f1.y; v[6]  = (__bf16)f1.z; v[7]  = (__bf16)f1.w;
    v[8]  = (__bf16)f2.x; v[9]  = (__bf16)f2.y; v[10] = (__bf16)f2.z; v[11] = (__bf16)f2.w;
    v[12] = (__bf16)f3.x; v[13] = (__bf16)f3.y; v[14] = (__bf16)f3.z; v[15] = (__bf16)f3.w;
    return v;
}

// t = x @ Wt + bt   ([N,64] @ [64,4])
__global__ __launch_bounds__(256) void na_tlogits_kernel(
        const float* __restrict__ x, const float* __restrict__ Wt,
        const float* __restrict__ bt, float* __restrict__ tOut, int N) {
    __shared__ __align__(16) float sW[260];
    int tid = threadIdx.x;
    for (int i = tid; i < 260; i += 256) sW[i] = (i < 256) ? Wt[i] : bt[i - 256];
    __syncthreads();
    int n = blockIdx.x * 256 + tid;
    if (n >= N) return;
    float4 acc = *(const float4*)&sW[256];
    const float* xr = x + (size_t)n * 64;
#pragma unroll
    for (int c4 = 0; c4 < 16; ++c4) {
        float4 xv = *(const float4*)(xr + c4 * 4);
#pragma unroll
        for (int j = 0; j < 4; ++j) {
            float  xs = f4get(xv, j);
            float4 wv = *(const float4*)&sW[(c4 * 4 + j) * 4];
            acc.x += xs * wv.x; acc.y += xs * wv.y;
            acc.z += xs * wv.z; acc.w += xs * wv.w;
        }
    }
    *(float4*)(tOut + (size_t)n * 4) = acc;
}

__global__ __launch_bounds__(256) void na_main_kernel(
        const float* __restrict__ src, const int* __restrict__ nidx,
        const float* __restrict__ Ws, const float* __restrict__ bs,
        const float* __restrict__ W_lin, const float* __restrict__ b_lin,
        const float* __restrict__ tbuf, float* __restrict__ out, int nNodes) {
    // B operands pre-swizzled to per-lane WMMA layout: [op][lane][8 dwords of bf16x2]
    __shared__ __align__(32) unsigned Bsw[NB_OPS * 32 * 8];   // 34 KB
    __shared__ float blin_s[256];                             // b_lin [4][64]
    __shared__ __align__(16) float asoft[WAVES][16][4];       // per-wave logits / weights

    const int tid = threadIdx.x;

    // ---- Build swizzled B (bf16) in LDS: cols 0..255 = W_lin[s][c][o], 256..259 = Ws ----
    for (int slot = tid; slot < NB_OPS * 32; slot += 256) {
        const int op    = slot >> 5;
        const int ln    = slot & 31;
        const int tn    = op >> 1;          // column tile
        const int kstep = op & 1;           // K half (0..31 / 32..63)
        const int gc    = tn * 16 + (ln & 15);
        const int kbase = kstep * 32 + (ln >> 4) * 8;   // B layout: lanes0-15 K=kb..kb+7, kb+16..kb+23
        unsigned* dst = &Bsw[slot * 8];
#pragma unroll
        for (int j = 0; j < 8; ++j) {
            const int k0 = kbase + 2 * j + (j >= 4 ? 8 : 0), k1 = k0 + 1;
            float f0 = 0.f, f1 = 0.f;
            if (gc < 256) {
                const int s = gc >> 6, o = gc & 63;
                f0 = W_lin[(s * 64 + k0) * 64 + o];
                f1 = W_lin[(s * 64 + k1) * 64 + o];
            } else if (gc < 260) {
                const int c2 = gc - 256;
                f0 = Ws[k0 * 4 + c2];
                f1 = Ws[k1 * 4 + c2];
            }
            dst[j] = pack_bf16(f0, f1);
        }
    }
    blin_s[tid] = b_lin[tid];
    __syncthreads();

    const int wave = tid >> 5, lane = tid & 31;
    const int half = lane >> 4, r16 = lane & 15;
    const int stride = gridDim.x * WAVES;

    for (int node = blockIdx.x * WAVES + wave; node < nNodes; node += stride) {
        // ---- Gather A operands (this node's 16 neighbor rows, 64 channels) ----
        const int    idxr = nidx[node * 16 + r16];
        const float* rowp = src + (size_t)idxr * 64;
        v16bf A0 = make_a(rowp, 0 * 32 + half * 8);
        v16bf A1 = make_a(rowp, 1 * 32 + half * 8);

        // ---- Assignment logits: msgs @ Ws  (column tile 16) ----
        v8f Ct = {};
        Ct = wmma_bf16(A0, *(const v16bf*)&Bsw[(32 * 32 + lane) * 8], Ct);
        Ct = wmma_bf16(A1, *(const v16bf*)&Bsw[(33 * 32 + lane) * 8], Ct);

        if (r16 < 4) {
#pragma unroll
            for (int v = 0; v < 8; ++v) asoft[wave][half * 8 + v][r16] = Ct[v];
        }
        __asm__ volatile("s_wait_dscnt 0" ::: "memory");   // wave-local LDS fence

        // ---- Softmax over S=4 (lanes 0..15, one row each); fold 1/size into weights ----
        if (lane < 16) {
            float4 lg  = *(const float4*)&asoft[wave][lane][0];
            float4 tv  = *(const float4*)(tbuf + (size_t)node * 4);
            float4 bsv = *(const float4*)bs;
            float l0 = lg.x + tv.x + bsv.x, l1 = lg.y + tv.y + bsv.y;
            float l2 = lg.z + tv.z + bsv.z, l3 = lg.w + tv.w + bsv.w;
            float m  = fmaxf(fmaxf(l0, l1), fmaxf(l2, l3));
            float e0 = __expf(l0 - m), e1 = __expf(l1 - m);
            float e2 = __expf(l2 - m), e3 = __expf(l3 - m);
            float inv = 0.25f / (e0 + e1 + e2 + e3);       // includes /size
            float4 av = {e0 * inv, e1 * inv, e2 * inv, e3 * inv};
            *(float4*)&asoft[wave][lane][0] = av;
        }
        __asm__ volatile("s_wait_dscnt 0" ::: "memory");

        float4 av[8];
#pragma unroll
        for (int v = 0; v < 8; ++v) av[v] = *(const float4*)&asoft[wave][half * 8 + v][0];

        const size_t outBase = ((size_t)node * 16 + half * 8) * 64;

        // ---- Expert matmuls + assignment-weighted combine ----
#pragma unroll
        for (int ob = 0; ob < 4; ++ob) {                   // output 16-col block
            float acc[8] = {0.f, 0.f, 0.f, 0.f, 0.f, 0.f, 0.f, 0.f};
#pragma unroll
            for (int s = 0; s < 4; ++s) {                  // expert
                const int op = (s * 4 + ob) * 2;
                v8f C = {};
                C = wmma_bf16(A0, *(const v16bf*)&Bsw[((op + 0) * 32 + lane) * 8], C);
                C = wmma_bf16(A1, *(const v16bf*)&Bsw[((op + 1) * 32 + lane) * 8], C);
                const float bl = blin_s[s * 64 + ob * 16 + r16];
#pragma unroll
                for (int v = 0; v < 8; ++v)
                    acc[v] += f4get(av[v], s) * (C[v] + bl);
            }
            const int o = ob * 16 + r16;
#pragma unroll
            for (int v = 0; v < 8; ++v)
                out[outBase + (size_t)v * 64 + o] = acc[v];
        }
    }
}

extern "C" void kernel_launch(void* const* d_in, const int* in_sizes, int n_in,
                              void* d_out, int out_size, void* d_ws, size_t ws_size,
                              hipStream_t stream) {
    const float* x     = (const float*)d_in[0];
    const float* src   = (const float*)d_in[1];
    const int*   nidx  = (const int*)  d_in[2];
    const float* Wt    = (const float*)d_in[3];
    const float* bt    = (const float*)d_in[4];
    const float* Ws    = (const float*)d_in[5];
    const float* bs    = (const float*)d_in[6];
    const float* W_lin = (const float*)d_in[7];
    const float* b_lin = (const float*)d_in[8];
    float*       out   = (float*)d_out;
    float*       tbuf  = (float*)d_ws;     // N*4 floats

    const int N = in_sizes[0] / 64;        // x is [N, 64]
    if (N <= 0) return;

    const int tb = (N + 255) / 256;
    hipLaunchKernelGGL(na_tlogits_kernel, dim3(tb), dim3(256), 0, stream,
                       x, Wt, bt, tbuf, N);

    int blocks = (N + WAVES - 1) / WAVES;
    if (blocks > 1024) blocks = 1024;
    hipLaunchKernelGGL(na_main_kernel, dim3(blocks), dim3(256), 0, stream,
                       src, nidx, Ws, bs, W_lin, b_lin, tbuf, out, N);
}
